// Seq2Seq_18811956756622
// MI455X (gfx1250) — matrix operand order repaired and checked by reading, not compiled, not run on hardware
//
#include <hip/hip_runtime.h>
#include <hip/hip_bf16.h>
#include <math.h>

// ---------------------------------------------------------------------------
// Seq2Seq GRU (L=2, H=D=512, B=16, V=32000, SRC=128, MAXLEN=64) for gfx1250.
// bf16 WMMA (v_wmma_f32_16x16x32_bf16) with fp32 accumulation.
// Weights are pre-converted to bf16 in workspace when ws_size allows
// (halves L2 traffic, removes per-WMMA cvt VALU); falls back to on-the-fly
// fp32->bf16 conversion otherwise.
// ---------------------------------------------------------------------------

typedef __bf16 bf16;
typedef __attribute__((ext_vector_type(16))) __bf16 v16bf;
typedef __attribute__((ext_vector_type(8)))  float  v8f;

#define HD 512
#define BT 16
#define VV 32000

struct alignas(16) BF8 { bf16 v[8]; };
struct alignas(8)  BF4 { bf16 v[4]; };

// ---- B fragment: lane l (l<16): col n = nb + l, K = kb..kb+15 (contiguous)
//      lane l (>=16): col n = nb + l-16, K = kb+16..kb+31

// fp32 source: 4 aligned float4 loads + cvt to bf16
__device__ inline v16bf b_frag(const float* __restrict__ W, int n, int kofs) {
  const float* p = W + (size_t)n * HD + kofs;
  float4 a = *(const float4*)(p + 0);
  float4 b = *(const float4*)(p + 4);
  float4 c = *(const float4*)(p + 8);
  float4 d = *(const float4*)(p + 12);
  v16bf r;
  r[0]=(bf16)a.x;  r[1]=(bf16)a.y;  r[2]=(bf16)a.z;  r[3]=(bf16)a.w;
  r[4]=(bf16)b.x;  r[5]=(bf16)b.y;  r[6]=(bf16)b.z;  r[7]=(bf16)b.w;
  r[8]=(bf16)c.x;  r[9]=(bf16)c.y;  r[10]=(bf16)c.z; r[11]=(bf16)c.w;
  r[12]=(bf16)d.x; r[13]=(bf16)d.y; r[14]=(bf16)d.z; r[15]=(bf16)d.w;
  return r;
}

// bf16 source: 2 aligned b128 loads, no conversion
__device__ inline v16bf b_frag(const bf16* __restrict__ W, int n, int kofs) {
  const BF8* p = (const BF8*)(W + (size_t)n * HD + kofs);
  BF8 u0 = p[0];
  BF8 u1 = p[1];
  v16bf r;
#pragma unroll
  for (int i = 0; i < 8; ++i) { r[i] = u0.v[i]; r[8 + i] = u1.v[i]; }
  return r;
}

// ---- A fragment (16x32 bf16) from LDS-resident activations [16][512]:
//      lane l<16 : row m=l,   K = kb..kb+15
//      lane l>=16: row m=l-16,K = kb+16..kb+31
__device__ inline v16bf a_frag_lds(const bf16* __restrict__ Xs, int lane, int kb) {
  int m  = lane & 15;
  int hi = lane >> 4;
  const BF8* p0 = (const BF8*)(Xs + m * HD + kb + hi * 16);
  const BF8* p1 = (const BF8*)(Xs + m * HD + kb + hi * 16 + 8);
  BF8 u0 = *p0;
  BF8 u1 = *p1;
  v16bf r;
#pragma unroll
  for (int i = 0; i < 8; ++i) { r[i] = u0.v[i]; r[8 + i] = u1.v[i]; }
  return r;
}

__device__ inline v8f vzero8() {
  v8f z = {0.f,0.f,0.f,0.f,0.f,0.f,0.f,0.f};
  return z;
}

// ---------------------------------------------------------------------------
// Fused GRU cell: one block per 16-wide hidden tile (32 blocks), 8 waves
// split K=512; six 16x16 WMMA gate tiles + gate nonlinearity + h update.
// ---------------------------------------------------------------------------
template <typename WT>
__global__ __launch_bounds__(256) void gru_cell_kernel(
    const float* __restrict__ xf32,   // [16,512] fp32 input (or null)
    const bf16*  __restrict__ xb16,   // [16,512] bf16 input (or null)
    const bf16*  __restrict__ hcur,   // [16,512] bf16 previous hidden
    bf16*        __restrict__ hnext,  // [16,512] bf16 new hidden
    const WT*    __restrict__ Wi,     // [3*512,512]
    const WT*    __restrict__ Wh,     // [3*512,512]
    const float* __restrict__ bi,     // [3*512]
    const float* __restrict__ bh)     // [3*512]
{
  __shared__ bf16 xs[BT * HD];
  __shared__ bf16 hs[BT * HD];
  __shared__ float red[7][6][256];

  const int tid  = threadIdx.x;
  const int lane = tid & 31;
  const int wave = tid >> 5;
  const int tile = blockIdx.x;            // hidden tile 0..31

  // ---- stage activations into LDS (bf16) ----
  if (xf32) {
    for (int i = tid; i < BT * HD; i += 256) xs[i] = (bf16)xf32[i];
  } else {
    const uint4* s = (const uint4*)xb16;
    uint4* d = (uint4*)xs;
    for (int i = tid; i < BT * HD / 8; i += 256) d[i] = s[i];
  }
  {
    const uint4* s = (const uint4*)hcur;
    uint4* d = (uint4*)hs;
    for (int i = tid; i < BT * HD / 8; i += 256) d[i] = s[i];
  }
  __syncthreads();

  v8f acc[6];
#pragma unroll
  for (int g = 0; g < 6; ++g) acc[g] = vzero8();

  const int ncol = tile * 16 + (lane & 15);
  const int kw0  = wave * 64;            // each wave: 64 of K=512 (2 k-tiles)

  for (int kb = kw0; kb < kw0 + 64; kb += 32) {
    v16bf ax = a_frag_lds(xs, lane, kb);
    v16bf ah = a_frag_lds(hs, lane, kb);
    const int kofs = kb + ((lane >> 4) ? 16 : 0);
#pragma unroll
    for (int g = 0; g < 3; ++g) {
      const int n = g * HD + ncol;
      __builtin_prefetch(Wi + (size_t)n * HD + kofs + 64, 0, 1);
      v16bf bwi = b_frag(Wi, n, kofs);
      acc[g] = __builtin_amdgcn_wmma_f32_16x16x32_bf16(
          false, ax, false, bwi, (short)0, acc[g], false, false);
      v16bf bwh = b_frag(Wh, n, kofs);
      acc[3 + g] = __builtin_amdgcn_wmma_f32_16x16x32_bf16(
          false, ah, false, bwh, (short)0, acc[3 + g], false, false);
    }
  }

  // ---- cross-wave K reduction via LDS ----
  if (wave > 0) {
#pragma unroll
    for (int g = 0; g < 6; ++g)
#pragma unroll
      for (int j = 0; j < 8; ++j)
        red[wave - 1][g][j * 32 + lane] = acc[g][j];
  }
  __syncthreads();

  if (wave == 0) {
#pragma unroll
    for (int g = 0; g < 6; ++g)
#pragma unroll
      for (int j = 0; j < 8; ++j) {
        float s = acc[g][j];
#pragma unroll
        for (int w = 0; w < 7; ++w) s += red[w][g][j * 32 + lane];
        acc[g][j] = s;
      }

    // ---- GRU gate math; C layout: vgpr j,lane l -> m = j + 8*(l>>4), n = l&15
    const float bir = bi[0 * HD + ncol], biz = bi[1 * HD + ncol], bin = bi[2 * HD + ncol];
    const float bhr = bh[0 * HD + ncol], bhz = bh[1 * HD + ncol], bhn = bh[2 * HD + ncol];
    const int half = lane >> 4;
#pragma unroll
    for (int j = 0; j < 8; ++j) {
      const int m = j + half * 8;
      float ir  = acc[0][j] + bir, iz = acc[1][j] + biz, inn = acc[2][j] + bin;
      float hr  = acc[3][j] + bhr, hz = acc[4][j] + bhz, hn  = acc[5][j] + bhn;
      float r   = 1.f / (1.f + __expf(-(ir + hr)));
      float z   = 1.f / (1.f + __expf(-(iz + hz)));
      float nn  = tanhf(inn + r * hn);
      float ho  = (float)hs[m * HD + ncol];
      hnext[m * HD + ncol] = (bf16)((1.f - z) * nn + z * ho);
    }
  }
}

// ---------------------------------------------------------------------------
// FC logits: 2000 vocab tiles of 16; 4 waves per block, one tile per wave,
// full K=512 (16 WMMA each). Output fp32 logits [16][32000].
// ---------------------------------------------------------------------------
template <typename WT>
__global__ __launch_bounds__(128) void fc_kernel(
    const bf16*  __restrict__ top,    // [16,512] decoder top hidden
    const WT*    __restrict__ fcW,    // [32000,512]
    const float* __restrict__ fcb,    // [32000]
    float*       __restrict__ logits) // [16,32000]
{
  __shared__ bf16 xs[BT * HD];
  const int tid = threadIdx.x, lane = tid & 31, wave = tid >> 5;
  {
    const uint4* s = (const uint4*)top;
    uint4* d = (uint4*)xs;
    for (int i = tid; i < BT * HD / 8; i += 128) d[i] = s[i];
  }
  __syncthreads();

  const int nt   = blockIdx.x * 4 + wave;   // 0..1999
  const int ncol = nt * 16 + (lane & 15);

  v8f acc = vzero8();
  for (int kb = 0; kb < HD; kb += 32) {
    v16bf a = a_frag_lds(xs, lane, kb);
    const int kofs = kb + ((lane >> 4) ? 16 : 0);
    __builtin_prefetch(fcW + (size_t)ncol * HD + kofs + 64, 0, 1);
    v16bf b = b_frag(fcW, ncol, kofs);
    acc = __builtin_amdgcn_wmma_f32_16x16x32_bf16(
        false, a, false, b, (short)0, acc, false, false);
  }

  const float bias = fcb[ncol];
  const int half = lane >> 4;
#pragma unroll
  for (int j = 0; j < 8; ++j) {
    const int m = j + half * 8;
    logits[(size_t)m * VV + ncol] = acc[j] + bias;
  }
}

// ---------------------------------------------------------------------------
// Per-batch-row log-softmax + argmax + teacher-forcing token select.
// ---------------------------------------------------------------------------
__global__ __launch_bounds__(256) void softmax_kernel(
    const float* __restrict__ logits,  // [16,32000]
    float*       __restrict__ outstep, // [16,32000] slice of d_out
    const int*   __restrict__ tgt,     // output_seq row t  [16]
    const int*   __restrict__ tf,      // tf_mask + t       [1]
    int*         __restrict__ tok)     // next-token state  [16]
{
  __shared__ float smax[256];
  __shared__ int   sidx[256];
  __shared__ float ssum[256];

  const int b = blockIdx.x, t = threadIdx.x;
  const float* row = logits + (size_t)b * VV;

  // max + first-argmax
  float mx = -INFINITY; int mi = 0;
  for (int v = t; v < VV; v += 256) {
    float x = row[v];
    if (x > mx) { mx = x; mi = v; }
  }
  smax[t] = mx; sidx[t] = mi;
  __syncthreads();
  for (int s = 128; s > 0; s >>= 1) {
    if (t < s) {
      float xo = smax[t + s]; int io = sidx[t + s];
      if (xo > smax[t] || (xo == smax[t] && io < sidx[t])) { smax[t] = xo; sidx[t] = io; }
    }
    __syncthreads();
  }
  const float rowmax = smax[0];
  const int   argmx  = sidx[0];

  // sum of exp
  float sum = 0.f;
  for (int v = t; v < VV; v += 256) sum += __expf(row[v] - rowmax);
  ssum[t] = sum;
  __syncthreads();
  for (int s = 128; s > 0; s >>= 1) {
    if (t < s) ssum[t] += ssum[t + s];
    __syncthreads();
  }
  const float logz = rowmax + logf(ssum[0]);

  for (int v = t; v < VV; v += 256)
    outstep[(size_t)b * VV + v] = row[v] - logz;

  if (t == 0) tok[b] = tf[0] ? tgt[b] : argmx;
}

// ---------------------------------------------------------------------------
// Embedding gather (fp32 -> bf16).
// ---------------------------------------------------------------------------
__global__ __launch_bounds__(128) void embed_kernel(
    const float* __restrict__ embed, const int* __restrict__ tok,
    bf16* __restrict__ xbuf)
{
  const int b = blockIdx.x;
  const float* src = embed + (size_t)tok[b] * HD;
  for (int i = threadIdx.x; i < HD; i += 128) xbuf[b * HD + i] = (bf16)src[i];
}

// ---------------------------------------------------------------------------
// One-shot fp32 -> bf16 weight conversion (vectorized 4-wide).
// ---------------------------------------------------------------------------
__global__ __launch_bounds__(256) void cvt_kernel(
    const float* __restrict__ s, bf16* __restrict__ d, int n4)
{
  for (int i = blockIdx.x * blockDim.x + threadIdx.x; i < n4;
       i += gridDim.x * blockDim.x) {
    float4 f = ((const float4*)s)[i];
    BF4 o;
    o.v[0] = (bf16)f.x; o.v[1] = (bf16)f.y;
    o.v[2] = (bf16)f.z; o.v[3] = (bf16)f.w;
    ((BF4*)d)[i] = o;
  }
}

// ---------------------------------------------------------------------------
// State init: zero hidden ping-pong buffers, SOS tokens, out row 0.
// ---------------------------------------------------------------------------
__global__ __launch_bounds__(256) void init_kernel(
    bf16* __restrict__ hslab, int* __restrict__ tok, float* __restrict__ out0)
{
  const int i = blockIdx.x * blockDim.x + threadIdx.x;
  if (i < 4 * BT * HD) hslab[i] = (bf16)0.f;
  if (i < 16) tok[i] = 0;
  for (int k = i; k < BT * VV; k += gridDim.x * blockDim.x) out0[k] = 0.f;
}

// ---------------------------------------------------------------------------
// Full pipeline over either fp32 or pre-converted bf16 weights.
// ---------------------------------------------------------------------------
template <typename WT>
static void run_pipeline(hipStream_t stream,
                         const float* input_seq, const int* output_seq,
                         const int* tf_mask, const float* embed,
                         const WT* encWi, const WT* encWh,
                         const float* encbi, const float* encbh,
                         const WT* decWi, const WT* decWh,
                         const float* decbi, const float* decbh,
                         const WT* fcW, const float* fcb,
                         float* out, bf16* h0a, bf16* h0b, bf16* h1a, bf16* h1b,
                         bf16* xbuf, float* logits, int* tok)
{
  const size_t WL = (size_t)3 * HD * HD;   // per-layer weight stride
  const size_t BL = (size_t)3 * HD;        // per-layer bias stride

  bf16 *c0 = h0a, *n0 = h0b, *c1 = h1a, *n1 = h1b, *tmp;

  // ---- encoder: 128 sequential steps x 2 layers ----
  for (int t = 0; t < 128; ++t) {
    gru_cell_kernel<WT><<<32, 256, 0, stream>>>(
        input_seq + (size_t)t * BT * HD, nullptr, c0, n0,
        encWi, encWh, encbi, encbh);
    gru_cell_kernel<WT><<<32, 256, 0, stream>>>(
        nullptr, n0, c1, n1,
        encWi + WL, encWh + WL, encbi + BL, encbh + BL);
    tmp = c0; c0 = n0; n0 = tmp;
    tmp = c1; c1 = n1; n1 = tmp;
  }

  // ---- decoder: 63 sequential steps ----
  for (int t = 1; t < 64; ++t) {
    embed_kernel<<<16, 128, 0, stream>>>(embed, tok, xbuf);
    gru_cell_kernel<WT><<<32, 256, 0, stream>>>(
        nullptr, xbuf, c0, n0, decWi, decWh, decbi, decbh);
    gru_cell_kernel<WT><<<32, 256, 0, stream>>>(
        nullptr, n0, c1, n1,
        decWi + WL, decWh + WL, decbi + BL, decbh + BL);
    tmp = c0; c0 = n0; n0 = tmp;
    tmp = c1; c1 = n1; n1 = tmp;
    fc_kernel<WT><<<500, 128, 0, stream>>>(c1, fcW, fcb, logits);
    softmax_kernel<<<16, 256, 0, stream>>>(
        logits, out + (size_t)t * BT * VV, output_seq + t * BT, tf_mask + t, tok);
  }
}

// ---------------------------------------------------------------------------
extern "C" void kernel_launch(void* const* d_in, const int* in_sizes, int n_in,
                              void* d_out, int out_size, void* d_ws, size_t ws_size,
                              hipStream_t stream) {
  const float* input_seq  = (const float*)d_in[0];   // [128,16,512]
  const int*   output_seq = (const int*)d_in[1];     // [64,16]
  const int*   tf_mask    = (const int*)d_in[2];     // [64]
  const float* embed      = (const float*)d_in[3];   // [32000,512]
  const float* encWi      = (const float*)d_in[4];   // [2,1536,512]
  const float* encWh      = (const float*)d_in[5];
  const float* encbi      = (const float*)d_in[6];   // [2,1536]
  const float* encbh      = (const float*)d_in[7];
  const float* decWi      = (const float*)d_in[8];
  const float* decWh      = (const float*)d_in[9];
  const float* decbi      = (const float*)d_in[10];
  const float* decbh      = (const float*)d_in[11];
  const float* fcW        = (const float*)d_in[12];  // [32000,512]
  const float* fcb        = (const float*)d_in[13];  // [32000]
  float* out = (float*)d_out;                        // [64,16,32000]

  // ---- workspace layout ----
  char* ws = (char*)d_ws;
  const size_t OFF_H      = 0;                               // 4 x [16][512] bf16
  const size_t OFF_X      = 65536;                           // [16][512] bf16
  const size_t OFF_TOK    = 81920;                           // [16] int (+pad)
  const size_t OFF_LOGITS = 82176;                           // [16][32000] f32
  const size_t OFF_W      = OFF_LOGITS + (size_t)BT * VV * 4; // bf16 weight pool

  bf16* hslab = (bf16*)(ws + OFF_H);
  bf16* h0a = hslab;
  bf16* h0b = hslab + 1 * BT * HD;
  bf16* h1a = hslab + 2 * BT * HD;
  bf16* h1b = hslab + 3 * BT * HD;
  bf16*  xbuf   = (bf16*)(ws + OFF_X);
  int*   tok    = (int*)  (ws + OFF_TOK);
  float* logits = (float*)(ws + OFF_LOGITS);

  const size_t N_GRU = (size_t)2 * 3 * HD * HD;   // elems per [2,1536,512]
  const size_t N_FC  = (size_t)VV * HD;           // elems of fc_W
  const size_t needW = OFF_W + (4 * N_GRU + N_FC) * sizeof(bf16);
  const bool   wbf   = (ws_size >= needW);

  init_kernel<<<512, 256, 0, stream>>>(hslab, tok, out);

  if (wbf) {
    bf16* pool    = (bf16*)(ws + OFF_W);
    bf16* encWi_b = pool;
    bf16* encWh_b = encWi_b + N_GRU;
    bf16* decWi_b = encWh_b + N_GRU;
    bf16* decWh_b = decWi_b + N_GRU;
    bf16* fcW_b   = decWh_b + N_GRU;

    cvt_kernel<<<1024, 256, 0, stream>>>(encWi, encWi_b, (int)(N_GRU / 4));
    cvt_kernel<<<1024, 256, 0, stream>>>(encWh, encWh_b, (int)(N_GRU / 4));
    cvt_kernel<<<1024, 256, 0, stream>>>(decWi, decWi_b, (int)(N_GRU / 4));
    cvt_kernel<<<1024, 256, 0, stream>>>(decWh, decWh_b, (int)(N_GRU / 4));
    cvt_kernel<<<2048, 256, 0, stream>>>(fcW,   fcW_b,   (int)(N_FC / 4));

    run_pipeline<bf16>(stream, input_seq, output_seq, tf_mask, embed,
                       encWi_b, encWh_b, encbi, encbh,
                       decWi_b, decWh_b, decbi, decbh,
                       fcW_b, fcb, out,
                       h0a, h0b, h1a, h1b, xbuf, logits, tok);
  } else {
    run_pipeline<float>(stream, input_seq, output_seq, tf_mask, embed,
                        encWi, encWh, encbi, encbh,
                        decWi, decWh, decbi, decbh,
                        fcW, fcb, out,
                        h0a, h0b, h1a, h1b, xbuf, logits, tok);
  }
}
